// Nonlocal_2156073583000
// MI455X (gfx1250) — matrix operand order repaired and checked by reading, not compiled, not run on hardware
//
#include <hip/hip_runtime.h>

// ---------------------------------------------------------------------------
// Non-local block (video self-attention) for MI455X / gfx1250, wave32 + WMMA.
// - all GEMMs via v_wmma_f32_16x16x32_bf16 (bf16 in, f32 accumulate)
// - LDS layouts make every WMMA fragment two ds_load_b128
// - double-buffered LDS + register staging: global loads for tile k+1 overlap
//   WMMA compute of tile k; one barrier per K-step
// - GEMM outputs feeding WMMA consumers are stored transposed (one 16B store
//   per lane from the D-tile layout), so consumers read contiguous fragments
// ---------------------------------------------------------------------------

typedef __attribute__((ext_vector_type(16))) __bf16 v16bf;
typedef __attribute__((ext_vector_type(8)))  __bf16 v8bf;
typedef __attribute__((ext_vector_type(8)))  float  v8f;

#define NB   8
#define CIN  1024
#define DI   512
#define THW  6272   // 8*28*28
#define PP   1568   // 8*14*14

// WMMA fragment maps (CDNA5 ISA 7.12.2, wave32), half = lane>>4, rc = lane&15:
//  A 16x32 bf16 (row=rc):  K = half*8 + i (i<8), 16 + half*8 + i-8 (i>=8)
//  B 32x16 bf16 (col=rc):  K = half*8 + i (i<8), 16 + half*8 + i-8 (i>=8)
//  C/D 16x16 f32 (col=rc): row = r + 8*half  -> 8 consecutive rows per lane

__device__ __forceinline__ v16bf combine16(v8bf lo, v8bf hi) {
  return __builtin_shufflevector(lo, hi, 0, 1, 2, 3, 4, 5, 6, 7,
                                 8, 9, 10, 11, 12, 13, 14, 15);
}

// ------------------------------------------------------- convert f32->bf16 --
__global__ void cvt_kernel(const float* __restrict__ s, __bf16* __restrict__ d, int n) {
  int i = blockIdx.x * 256 + threadIdx.x;
  if (i < n) d[i] = (__bf16)s[i];
}

// ----------------------------------------------- maxpool 1x2x2 + cast bf16 --
__global__ void maxpool_kernel(const float* __restrict__ x, __bf16* __restrict__ xp, long total) {
  long i = (long)blockIdx.x * 256 + threadIdx.x;
  if (i >= total) return;
  long hw = i % 196;           // 14*14
  long rest = i / 196;         // (n*1024 + c)*8 + t
  long yy = hw / 14, xx = hw % 14;
  const float* src = x + rest * 784 + yy * 56 + xx * 2;
  float v = fmaxf(fmaxf(src[0], src[1]), fmaxf(src[28], src[29]));
  xp[i] = (__bf16)v;
}

// ------------------------------------------------------- tiled WMMA GEMM ----
// D = A[M][K] * B (+bias). A bf16 row-major. B normal: [K][N]; TRANS_B: [N][K].
// TRANS_D: D stored [N][M] bf16. Block 256 = 8 waves; tile 128(M) x 64(N);
// wave w owns M-strip w*16 and all 4 N-tiles. Double-buffered pipeline.
#define BM 128
#define BN 64
#define BK 32
#define LDA 40   // row stride (elems): 80B = 16B-aligned, conflict-free banks
#define LDB 40

template <typename TIN, typename TOUT, bool TRANS_B, bool TRANS_D, bool HAS_BIAS>
__global__ __launch_bounds__(256) void gemm_wmma(
    const __bf16* __restrict__ A, const TIN* __restrict__ B,
    const float* __restrict__ bias, TOUT* __restrict__ D,
    int M, int N, int K, long strideA, long strideB, long strideD)
{
  __shared__ __align__(16) __bf16 sA[2][BM][LDA];   // [m][k] 2x10.2 KB
  __shared__ __align__(16) __bf16 sB[2][BN][LDB];   // [n][k] 2x5.1 KB (transposed)
  const int tid  = threadIdx.x;
  const int lane = tid & 31;
  const int wave = tid >> 5;
  const int half = lane >> 4;
  const int rc   = lane & 15;
  const int n0 = blockIdx.x * BN;
  const int m0 = blockIdx.y * BM;
  const long bA = (long)blockIdx.z * strideA;
  const long bB = (long)blockIdx.z * strideB;
  const long bD = (long)blockIdx.z * strideD;
  const bool fullN = (n0 + BN <= N);

  // register staging (double-buffer pipeline)
  const int mA  = tid >> 1;
  const int kkA = (tid & 1) * 16;
  const __bf16* rowA = A + bA + (long)(m0 + mA) * K + kkA;
  const int nnT = tid >> 2, kkT = (tid & 3) * 8;   // TRANS_B mapping
  const int kkN = tid >> 3, nbN = (tid & 7) * 8;   // normal mapping
  v8bf ra0, ra1, rbh; v8f rbf;

  auto loadA = [&](int k0) {
    ra0 = *(const v8bf*)(rowA + k0);
    ra1 = *(const v8bf*)(rowA + k0 + 8);
    if (k0 + 2 * BK < K) __builtin_prefetch(rowA + k0 + 2 * BK, 0, 1);
  };
  auto storeA = [&](int buf) {
    *(v8bf*)&sA[buf][mA][kkA]     = ra0;
    *(v8bf*)&sA[buf][mA][kkA + 8] = ra1;
  };
  auto loadB = [&](int k0) {
    if constexpr (TRANS_B) {
      const TIN* src = B + bB + (long)(n0 + nnT) * K + k0 + kkT;
      v8bf v = {};
      if (fullN || n0 + nnT < N) v = *(const v8bf*)src;
      rbh = v;
      if (k0 + 2 * BK < K) __builtin_prefetch(src + 2 * BK, 0, 1);
    } else {
      const TIN* src = B + bB + (long)(k0 + kkN) * N + n0 + nbN;
      if constexpr (sizeof(TIN) == 4) {
        if (fullN || n0 + nbN + 8 <= N) rbf = *(const v8f*)src;
        else {
          #pragma unroll
          for (int e = 0; e < 8; ++e) rbf[e] = (n0 + nbN + e < N) ? (float)src[e] : 0.0f;
        }
      } else {
        if (fullN || n0 + nbN + 8 <= N) rbh = *(const v8bf*)src;
        else {
          #pragma unroll
          for (int e = 0; e < 8; ++e) rbh[e] = (n0 + nbN + e < N) ? src[e] : (__bf16)0.0f;
        }
      }
      if (k0 + 2 * BK < K) __builtin_prefetch(src + (long)2 * BK * N, 0, 1);
    }
  };
  auto storeB = [&](int buf) {
    if constexpr (TRANS_B) {
      *(v8bf*)&sB[buf][nnT][kkT] = rbh;
    } else if constexpr (sizeof(TIN) == 4) {
      #pragma unroll
      for (int e = 0; e < 8; ++e) sB[buf][nbN + e][kkN] = (__bf16)rbf[e];
    } else {
      #pragma unroll
      for (int e = 0; e < 8; ++e) sB[buf][nbN + e][kkN] = rbh[e];
    }
  };

  v8f acc[4] = {{}, {}, {}, {}};
  loadA(0); loadB(0);
  int buf = 0;
  for (int k0 = 0; k0 < K; k0 += BK) {
    storeA(buf); storeB(buf);
    __syncthreads();                       // single barrier per K-step
    if (k0 + BK < K) { loadA(k0 + BK); loadB(k0 + BK); }  // overlap w/ compute

    const __bf16* pa = &sA[buf][wave * 16 + rc][half * 8];
    v16bf af = combine16(*(const v8bf*)pa, *(const v8bf*)(pa + 16));
    v16bf bfr[4];
    #pragma unroll
    for (int nt = 0; nt < 4; ++nt) {       // issue all LDS loads first
      const __bf16* pb = &sB[buf][nt * 16 + rc][half * 8];
      bfr[nt] = combine16(*(const v8bf*)pb, *(const v8bf*)(pb + 16));
    }
    #pragma unroll
    for (int nt = 0; nt < 4; ++nt)
      acc[nt] = __builtin_amdgcn_wmma_f32_16x16x32_bf16(
          false, af, false, bfr[nt], (short)0, acc[nt], false, false);
    buf ^= 1;
  }

  // ---- epilogue ----
  const int mrow = m0 + wave * 16 + half * 8;  // base of this lane's 8 rows
  if constexpr (TRANS_D) {
    #pragma unroll
    for (int nt = 0; nt < 4; ++nt) {
      int n = n0 + nt * 16 + rc;
      if (n < N) {
        v8bf vb;
        #pragma unroll
        for (int r = 0; r < 8; ++r) {
          float bv = HAS_BIAS ? bias[mrow + r] : 0.0f;
          vb[r] = (__bf16)(acc[nt][r] + bv);
        }
        *(v8bf*)&D[bD + (long)n * M + mrow] = vb;   // one 16B store per tile
      }
    }
  } else {
    #pragma unroll
    for (int r = 0; r < 8; ++r) {
      int m = mrow + r;
      float bv = HAS_BIAS ? bias[m] : 0.0f;
      #pragma unroll
      for (int nt = 0; nt < 4; ++nt) {
        int n = n0 + nt * 16 + rc;
        if (n < N) D[bD + (long)m * N + n] = (TOUT)(acc[nt][r] + bv);
      }
    }
  }
}

// ------------------------------------------------- attention (two passes) ---
// logits[t,p] = sum_c thetaT[t,c]*phiT[p,c] * DI^-0.5
// Pass 1: online row max m / sum-exp l -> mbuf/lbuf.
// Pass 2: probs = exp(logit-m)/l -> attn bf16 [t][p].
// 8 waves/block, one 16-query tile per wave. theta A-fragments persistent in
// VGPRs; phi key tiles double-buffered in LDS, staged through registers; the
// kc fragment loop is pipelined one-ahead under the WMMA chain.
template <bool WRITE_PROBS>
__global__ __launch_bounds__(256) void attn_kernel(
    const __bf16* __restrict__ thetaT,  // [nb][THW][DI]
    const __bf16* __restrict__ phiT,    // [nb][PP][DI]
    float* __restrict__ mbuf,           // [nb][THW]
    float* __restrict__ lbuf,           // [nb][THW]
    __bf16* __restrict__ attn)          // [nb][THW][PP]
{
  __shared__ __align__(16) __bf16 sPhi[2][16][DI + 8];   // [p][c], 2x16.6 KB
  const int tid  = threadIdx.x;
  const int lane = tid & 31;
  const int wave = tid >> 5;
  const int half = lane >> 4;
  const int rc   = lane & 15;
  const int nb   = blockIdx.y;
  const int t0   = (blockIdx.x * 8 + wave) * 16;
  const __bf16* thT = thetaT + (long)nb * THW * DI;
  const __bf16* phT = phiT   + (long)nb * PP * DI;

  // persistent theta A-fragments: contiguous vector loads from thetaT row
  v16bf af[16];
  {
    const __bf16* trow = thT + (long)(t0 + rc) * DI + half * 8;
    #pragma unroll
    for (int kc = 0; kc < 16; ++kc)
      af[kc] = combine16(*(const v8bf*)(trow + kc * 32),
                         *(const v8bf*)(trow + kc * 32 + 16));
  }

  float m8[8], l8[8];
  #pragma unroll
  for (int r = 0; r < 8; ++r) {
    if (WRITE_PROBS) {
      long row = (long)nb * THW + t0 + r + half * 8;
      m8[r] = mbuf[row];
      l8[r] = 1.0f / lbuf[row];
    } else {
      m8[r] = -3.0e38f;
      l8[r] = 0.0f;
    }
  }

  // register staging of one 16-key phi tile (64B contiguous per thread)
  const int pr = tid >> 4;
  const int cg = (tid & 15) * 32;
  const __bf16* srcPhi = phT + (long)pr * DI + cg;
  v8bf ph0, ph1, ph2, ph3;
  auto loadPhi = [&](int p0) {
    const __bf16* s = srcPhi + (long)p0 * DI;
    ph0 = *(const v8bf*)(s);
    ph1 = *(const v8bf*)(s + 8);
    ph2 = *(const v8bf*)(s + 16);
    ph3 = *(const v8bf*)(s + 24);
    if (p0 + 32 < PP) __builtin_prefetch(s + 32 * DI, 0, 1);
  };
  auto storePhi = [&](int buf) {
    *(v8bf*)&sPhi[buf][pr][cg]      = ph0;
    *(v8bf*)&sPhi[buf][pr][cg + 8]  = ph1;
    *(v8bf*)&sPhi[buf][pr][cg + 16] = ph2;
    *(v8bf*)&sPhi[buf][pr][cg + 24] = ph3;
  };

  const float scale = 0.04419417382415922f;  // 512^-0.5

  loadPhi(0);
  int buf = 0;
  for (int kt = 0; kt < PP / 16; ++kt) {
    const int p0 = kt * 16;
    storePhi(buf);
    __syncthreads();                        // single barrier per key tile
    if (p0 + 16 < PP) loadPhi(p0 + 16);     // overlap with compute below

    const __bf16* base = &sPhi[buf][rc][half * 8];
    v16bf bf = combine16(*(const v8bf*)base, *(const v8bf*)(base + 16));
    v8f lg = {};
    #pragma unroll
    for (int kc = 0; kc < 16; ++kc) {       // one-ahead pipelined fragments
      v16bf cur = bf;
      if (kc < 15) {
        const __bf16* pb = base + (kc + 1) * 32;
        bf = combine16(*(const v8bf*)pb, *(const v8bf*)(pb + 16));
      }
      lg = __builtin_amdgcn_wmma_f32_16x16x32_bf16(
          false, af[kc], false, cur, (short)0, lg, false, false);
    }

    #pragma unroll
    for (int r = 0; r < 8; ++r) {
      float v = lg[r] * scale;
      if (WRITE_PROBS) {
        float prob = __expf(v - m8[r]) * l8[r];
        attn[(long)nb * THW * PP + (long)(t0 + r + half * 8) * PP + p0 + rc] = (__bf16)prob;
      } else {
        float rm = v;   // row reductions across the 16 lanes holding the cols
        #pragma unroll
        for (int off = 8; off >= 1; off >>= 1)
          rm = fmaxf(rm, __shfl_xor(rm, off, 16));
        float m2 = fmaxf(m8[r], rm);
        float e  = __expf(v - m2);
        #pragma unroll
        for (int off = 8; off >= 1; off >>= 1)
          e += __shfl_xor(e, off, 16);
        l8[r] = l8[r] * __expf(m8[r] - m2) + e;
        m8[r] = m2;
      }
    }
    buf ^= 1;
  }

  if (!WRITE_PROBS && rc == 0) {
    #pragma unroll
    for (int r = 0; r < 8; ++r) {
      long row = (long)nb * THW + t0 + r + half * 8;
      mbuf[row] = m8[r];
      lbuf[row] = l8[r];
    }
  }
}

// -------------------------------------------------- batchnorm statistics ----
__global__ __launch_bounds__(256) void bn_stats_kernel(
    const float* __restrict__ p, float* __restrict__ mean, float* __restrict__ invstd)
{
  const int c = blockIdx.x;
  float s = 0.0f, s2 = 0.0f;
  const long per = (long)NB * THW;
  for (long i = threadIdx.x; i < per; i += 256) {
    long n = i / THW;
    long t = i - n * THW;
    float v = p[(n * CIN + c) * (long)THW + t];
    s += v; s2 += v * v;
  }
  __shared__ float rs[256], rq[256];
  rs[threadIdx.x] = s; rq[threadIdx.x] = s2;
  __syncthreads();
  for (int st = 128; st > 0; st >>= 1) {
    if (threadIdx.x < st) { rs[threadIdx.x] += rs[threadIdx.x + st]; rq[threadIdx.x] += rq[threadIdx.x + st]; }
    __syncthreads();
  }
  if (threadIdx.x == 0) {
    float inv = 1.0f / (float)per;
    float mu  = rs[0] * inv;
    float var = rq[0] * inv - mu * mu;
    mean[c]   = mu;
    invstd[c] = rsqrtf(var + 1e-5f);
  }
}

// ----------------------------------------------- bn apply + residual add ----
__global__ void final_kernel(
    const float* __restrict__ x, const float* __restrict__ p,
    const float* __restrict__ mean, const float* __restrict__ invstd,
    const float* __restrict__ gamma, const float* __restrict__ beta,
    float* __restrict__ y, long total)
{
  long i = (long)blockIdx.x * 256 + threadIdx.x;
  if (i >= total) return;
  long c = (i / THW) % CIN;
  y[i] = x[i] + (p[i] - mean[c]) * invstd[c] * gamma[c] + beta[c];
}

// --------------------------------------------------------------- launcher ---
extern "C" void kernel_launch(void* const* d_in, const int* in_sizes, int n_in,
                              void* d_out, int out_size, void* d_ws, size_t ws_size,
                              hipStream_t stream) {
  const float* x       = (const float*)d_in[0];
  const float* w_theta = (const float*)d_in[1];
  const float* b_theta = (const float*)d_in[2];
  const float* w_phi   = (const float*)d_in[3];
  const float* b_phi   = (const float*)d_in[4];
  const float* w_g     = (const float*)d_in[5];
  const float* b_g     = (const float*)d_in[6];
  const float* w_out   = (const float*)d_in[7];
  const float* b_out   = (const float*)d_in[8];
  const float* gamma   = (const float*)d_in[9];
  const float* beta    = (const float*)d_in[10];
  float* y = (float*)d_out;

  // --- workspace layout (p aliases theta+attn; total ~316 MB) ---
  char* ws = (char*)d_ws;
  size_t off = 0;
  auto take = [&](size_t bytes) -> void* {
    void* ptr = ws + off;
    off += (bytes + 255) & ~(size_t)255;
    return ptr;
  };
  __bf16* wthb   = (__bf16*)take((size_t)DI * CIN * 2);
  __bf16* wphb   = (__bf16*)take((size_t)DI * CIN * 2);
  __bf16* wgb    = (__bf16*)take((size_t)DI * CIN * 2);
  __bf16* woutb  = (__bf16*)take((size_t)CIN * DI * 2);
  __bf16* xp     = (__bf16*)take((size_t)NB * CIN * PP * 2);
  size_t  thoff  = off;
  __bf16* thetaT = (__bf16*)take((size_t)NB * THW * DI * 2);
  __bf16* attn   = (__bf16*)take((size_t)NB * THW * PP * 2);
  float*  pbuf   = (float*)(ws + thoff);   // aliases thetaT+attn (dead by then)
  __bf16* phiT   = (__bf16*)take((size_t)NB * PP * DI * 2);
  __bf16* g      = (__bf16*)take((size_t)NB * DI * PP * 2);
  __bf16* outb   = (__bf16*)take((size_t)NB * DI * THW * 2);
  float*  mbuf   = (float*)take((size_t)NB * THW * 4);
  float*  lbuf   = (float*)take((size_t)NB * THW * 4);
  float*  meanb  = (float*)take((size_t)CIN * 4);
  float*  invsb  = (float*)take((size_t)CIN * 4);

  // 1) weights -> bf16
  {
    int n = DI * CIN;
    cvt_kernel<<<(n + 255) / 256, 256, 0, stream>>>(w_theta, wthb, n);
    cvt_kernel<<<(n + 255) / 256, 256, 0, stream>>>(w_phi,   wphb, n);
    cvt_kernel<<<(n + 255) / 256, 256, 0, stream>>>(w_g,     wgb,  n);
    cvt_kernel<<<(n + 255) / 256, 256, 0, stream>>>(w_out,   woutb, n);
  }
  // 2) maxpool -> xp bf16
  {
    long n = (long)NB * CIN * PP;
    maxpool_kernel<<<(unsigned)((n + 255) / 256), 256, 0, stream>>>(x, xp, n);
  }
  // 3) thetaT[t][c] = (Wtheta @ x + b)^T   (TRANS_D)
  gemm_wmma<float, __bf16, false, true, true>
      <<<dim3(THW / BN, DI / BM, NB), 256, 0, stream>>>(
      wthb, x, b_theta, thetaT, DI, THW, CIN, 0L, (long)CIN * THW, (long)THW * DI);
  // 4) phiT[p][c] = (Wphi @ xp + b)^T      (TRANS_D, N masked: 1568)
  gemm_wmma<__bf16, __bf16, false, true, true>
      <<<dim3((PP + BN - 1) / BN, DI / BM, NB), 256, 0, stream>>>(
      wphb, xp, b_phi, phiT, DI, PP, CIN, 0L, (long)CIN * PP, (long)PP * DI);
  // 5) g[c][p] = Wg @ xp + b               (normal D: used as A later)
  gemm_wmma<__bf16, __bf16, false, false, true>
      <<<dim3((PP + BN - 1) / BN, DI / BM, NB), 256, 0, stream>>>(
      wgb, xp, b_g, g, DI, PP, CIN, 0L, (long)CIN * PP, (long)DI * PP);
  // 6) attention pass 1: softmax stats (m, l)
  attn_kernel<false><<<dim3(THW / (16 * 8), NB), 256, 0, stream>>>(thetaT, phiT, mbuf, lbuf, attn);
  // 7) attention pass 2: normalized probs -> attn bf16 [t][p]
  attn_kernel<true><<<dim3(THW / (16 * 8), NB), 256, 0, stream>>>(thetaT, phiT, mbuf, lbuf, attn);
  // 8) out[c][t] = g @ attn^T              (TRANS_B)
  gemm_wmma<__bf16, __bf16, true, false, false>
      <<<dim3(THW / BN, DI / BM, NB), 256, 0, stream>>>(
      g, attn, nullptr, outb, DI, THW, PP, (long)DI * PP, (long)THW * PP, (long)DI * THW);
  // 9) p = Wout @ out + b_out, f32 (aliases thetaT+attn)
  gemm_wmma<__bf16, float, false, false, true>
      <<<dim3(THW / BN, CIN / BM, NB), 256, 0, stream>>>(
      woutb, outb, b_out, pbuf, CIN, THW, DI, 0L, (long)DI * THW, (long)CIN * THW);
  // 10) batchnorm stats per channel
  bn_stats_kernel<<<CIN, 256, 0, stream>>>(pbuf, meanb, invsb);
  // 11) y = x + gamma*(p-mean)*invstd + beta
  {
    long n = (long)NB * CIN * THW;
    final_kernel<<<(unsigned)((n + 255) / 256), 256, 0, stream>>>(
        x, pbuf, meanb, invsb, gamma, beta, y, n);
  }
}
